// EqvTransformer_89550068122337
// MI455X (gfx1250) — matrix operand order repaired and checked by reading, not compiled, not run on hardware
//
#include <hip/hip_runtime.h>
#include <hip/hip_bf16.h>

// EqvTransformer for MI455X (gfx1250): bf16 WMMA everywhere, flash-style
// attention fused with the pairwise loc-MLP, X_pairs read exactly once and
// staged to LDS via the async global->LDS copy engine (ASYNCcnt).

typedef __attribute__((ext_vector_type(16))) __bf16 v16bf;
typedef __attribute__((ext_vector_type(8)))  __bf16 v8bf;
typedef __attribute__((ext_vector_type(8)))  float  v8f;

#define Bsz 4
#define Nsz 1024
#define Dsz 512
#define Hsz 8
#define HDsz 64
#define INFM 1e38f

__device__ __forceinline__ v8f wmma_bf16(v16bf a, v16bf b, v8f c) {
  // emits v_wmma_f32_16x16x32_bf16
  return __builtin_amdgcn_wmma_f32_16x16x32_bf16(false, a, false, b, (short)0, c, false, false);
}

// A operand, 16x32 (M x K), source row-major [row][col], ISA 7.12.2 layout:
// lane L: M = L%16; elems 0..7 -> K = (L/16)*8 + i ; elems 8..15 -> K = 16 + (L/16)*8 + i
__device__ __forceinline__ v16bf frag_a(const __bf16* src, int row0, int col0, int ld, int lane) {
  const int m = lane & 15, half = lane >> 4;
  const __bf16* p = src + (size_t)(row0 + m) * ld + col0 + half * 8;
  v8bf lo = *(const v8bf*)p;
  v8bf hi = *(const v8bf*)(p + 16);
  return __builtin_shufflevector(lo, hi, 0,1,2,3,4,5,6,7,8,9,10,11,12,13,14,15);
}

// B operand, 32x16 (K x N), source stored N-major: src[n][k] == B[k][n].
// lane L: N = L%16; elems 0..7 -> K = (L/16)*8 + i ; elems 8..15 -> K = 16 + (L/16)*8 + i
__device__ __forceinline__ v16bf frag_b(const __bf16* src, int n0, int k0, int ld, int lane) {
  const int n = lane & 15, kh = (lane >> 4) * 8;
  const __bf16* p = src + (size_t)(n0 + n) * ld + k0 + kh;
  v8bf lo = *(const v8bf*)p;
  v8bf hi = *(const v8bf*)(p + 16);
  return __builtin_shufflevector(lo, hi, 0,1,2,3,4,5,6,7,8,9,10,11,12,13,14,15);
}

// ---------------- kernel 1: fp32 -> bf16 conversions ----------------
__global__ __launch_bounds__(256) void prep_kernel(
    const float* __restrict__ Y,
    const float* __restrict__ Wq, const float* __restrict__ Wk,
    const float* __restrict__ Wv, const float* __restrict__ Wo,
    __bf16* __restrict__ Ybf, __bf16* __restrict__ Wqbf, __bf16* __restrict__ Wkbf,
    __bf16* __restrict__ Wvbf, __bf16* __restrict__ Wobf)
{
  const int i = blockIdx.x * 256 + threadIdx.x;
  if (i < Bsz * Nsz * Dsz) Ybf[i] = (__bf16)Y[i];
  if (i < Dsz * Dsz) {
    Wqbf[i] = (__bf16)Wq[i];
    Wkbf[i] = (__bf16)Wk[i];
    Wvbf[i] = (__bf16)Wv[i];
    Wobf[i] = (__bf16)Wo[i];
  }
}

// ---------------- kernel 2: QKV projections via WMMA ----------------
// grid (32, 32, 3): z selects Q/K/V; block = 8 waves, each wave one 16x16 tile.
__global__ __launch_bounds__(256) void qkv_kernel(
    const __bf16* __restrict__ Ybf,
    const __bf16* __restrict__ Wqbf, const __bf16* __restrict__ Wkbf,
    const __bf16* __restrict__ Wvbf,
    const float* __restrict__ bq, const float* __restrict__ bk,
    const float* __restrict__ bv,
    __bf16* __restrict__ Qbf, __bf16* __restrict__ Kbf,
    float* __restrict__ Vf32, __bf16* __restrict__ Vtbf)
{
  const int wave = threadIdx.x >> 5, lane = threadIdx.x & 31;
  const int m0 = blockIdx.x * 128 + wave * 16;
  const int n0 = blockIdx.y * 16;
  const int w  = blockIdx.z;
  const __bf16* Wb  = (w == 0) ? Wqbf : (w == 1) ? Wkbf : Wvbf;
  const float* bias = (w == 0) ? bq   : (w == 1) ? bk   : bv;

  v8f acc = {};
  for (int k = 0; k < Dsz; k += 32) {
    v16bf a  = frag_a(Ybf, m0, k, Dsz, lane);   // X tile
    v16bf bb = frag_b(Wb,  n0, k, Dsz, lane);   // W^T tile (W is N-major)
    acc = wmma_bf16(a, bb, acc);
  }

  const int mloc = lane & 15, half = lane >> 4;
  const int n  = n0 + mloc;
  const float bval = bias[n];
  const int hh = n >> 6, dd = n & 63;
#pragma unroll
  for (int r = 0; r < 8; ++r) {
    const int mm = m0 + r + 8 * half;
    const int bi = mm >> 10, q = mm & 1023;
    const float v = acc[r] + bval;
    const size_t hidx = (((size_t)(bi * Hsz + hh)) * Nsz + q) * HDsz + dd;
    if (w == 0)      Qbf[hidx] = (__bf16)v;
    else if (w == 1) Kbf[hidx] = (__bf16)v;
    else {
      Vf32[(size_t)mm * Dsz + n] = v;
      Vtbf[(((size_t)(bi * Hsz + hh)) * HDsz + dd) * Nsz + q] = (__bf16)v;  // V^T
    }
  }
}

// ---------------- kernel 3: fused attention + loc-MLP (flash style) ----------------
// grid (N/16, B); block = 256 = 8 waves = 8 heads sharing one X_pairs tile.
__global__ __launch_bounds__(256) void attn_kernel(
    const __bf16* __restrict__ Qbf, const __bf16* __restrict__ Kbf,
    const __bf16* __restrict__ Vtbf, const float* __restrict__ Vf32,
    const float* __restrict__ Xp, const float* __restrict__ pres,
    const float* __restrict__ W1, const float* __restrict__ b1,
    const float* __restrict__ W2, const float* __restrict__ b2,
    float* __restrict__ Ocur, __bf16* __restrict__ Ocurbf)
{
  const int b    = blockIdx.y;
  const int q0   = blockIdx.x * 16;
  const int h    = threadIdx.x >> 5;          // wave == head
  const int lane = threadIdx.x & 31;
  const int m    = lane & 15;
  const int half = lane >> 4;

  __shared__ float xtile[16][32][3];                       // shared by all 8 heads
  __shared__ __align__(16) __bf16 ptile[Hsz][16][32];      // per-wave P transpose scratch

  const __bf16* Qb = Qbf  + (size_t)(b * Hsz + h) * Nsz * HDsz;
  const __bf16* Kb = Kbf  + (size_t)(b * Hsz + h) * Nsz * HDsz;
  const __bf16* Vt = Vtbf + (size_t)(b * Hsz + h) * HDsz * Nsz;

  float w1[3][3], bb1[3], w2[3];
#pragma unroll
  for (int o = 0; o < 3; ++o) {
    bb1[o] = b1[h * 3 + o];
    w2[o]  = W2[h * 3 + o];
#pragma unroll
    for (int c = 0; c < 3; ++c) w1[o][c] = W1[h * 9 + o * 3 + c];
  }
  const float bb2 = b2[h];

  const v16bf qa0 = frag_a(Qb, q0, 0,  HDsz, lane);
  const v16bf qa1 = frag_a(Qb, q0, 32, HDsz, lane);

  float pq[8], rmax[8], rsum[8];
#pragma unroll
  for (int r = 0; r < 8; ++r) {
    pq[r]   = pres[b * Nsz + q0 + r + 8 * half];
    rmax[r] = -3.0e38f;
    rsum[r] = 0.f;
  }
  v8f oacc[4];
#pragma unroll
  for (int j = 0; j < 4; ++j) oacc[j] = v8f{};

  const float inv_sqrt = 0.04419417382415922f;   // 1/sqrt(512)

  for (int kc = 0; kc < Nsz; kc += 32) {
    __syncthreads();   // previous chunk's xtile reads done
    // Stage X_pairs tile with the async global->LDS copy engine (ASYNCcnt):
    // 1536 floats / 256 threads = 6 async b32 copies each, no VGPR round-trip.
#pragma unroll
    for (int it = 0; it < 6; ++it) {
      const int idx = it * 256 + threadIdx.x;
      const int qi = idx / 96, rr = idx - qi * 96;
      const int ki = rr / 3,  c  = rr - ki * 3;
      const unsigned goff  = (unsigned)(((((size_t)(b * Nsz + q0 + qi)) * Nsz
                                          + kc + ki) * 3 + c) * 4);
      const unsigned laddr = (unsigned)(size_t)&xtile[qi][ki][c];  // LDS addr = addr[31:0]
      asm volatile("global_load_async_to_lds_b32 %0, %1, %2"
                   :: "v"(laddr), "v"(goff), "s"(Xp) : "memory");
    }
    asm volatile("s_wait_asynccnt 0x0" ::: "memory");
    __syncthreads();
    if (kc + 32 < Nsz)  // global_prefetch_b8 for next X_pairs tile
      __builtin_prefetch(&Xp[(((size_t)(b * Nsz + q0 + (threadIdx.x >> 4))) * Nsz
                              + kc + 32 + ((threadIdx.x & 15) << 1)) * 3], 0, 0);

    // S = Q K^T for two 16-col sub-tiles (HD=64 -> 2 WMMAs each)
    v8f s0 = {}, s1 = {};
    {
      v16bf k00 = frag_b(Kb, kc,      0,  HDsz, lane);
      v16bf k01 = frag_b(Kb, kc,      32, HDsz, lane);
      s0 = wmma_bf16(qa0, k00, s0);
      s0 = wmma_bf16(qa1, k01, s0);
      v16bf k10 = frag_b(Kb, kc + 16, 0,  HDsz, lane);
      v16bf k11 = frag_b(Kb, kc + 16, 32, HDsz, lane);
      s1 = wmma_bf16(qa0, k10, s1);
      s1 = wmma_bf16(qa1, k11, s1);
    }

    const float pk0 = pres[b * Nsz + kc + m];
    const float pk1 = pres[b * Nsz + kc + 16 + m];

#pragma unroll
    for (int r = 0; r < 8; ++r) {
      const int qi = r + 8 * half;
      float lg[2];
#pragma unroll
      for (int t = 0; t < 2; ++t) {
        const float* x = xtile[qi][t * 16 + m];
        const float a0 = fmaxf(w1[0][0]*x[0] + w1[0][1]*x[1] + w1[0][2]*x[2] + bb1[0], 0.f);
        const float a1 = fmaxf(w1[1][0]*x[0] + w1[1][1]*x[1] + w1[1][2]*x[2] + bb1[1], 0.f);
        const float a2 = fmaxf(w1[2][0]*x[0] + w1[2][1]*x[1] + w1[2][2]*x[2] + bb1[2], 0.f);
        const float loc = a0*w2[0] + a1*w2[1] + a2*w2[2] + bb2;
        float v = (t ? s1[r] : s0[r]) * inv_sqrt + loc;
        const float pqr = pq[r];
        v = pqr * v - (1.f - pqr) * INFM;
        const float pkv = t ? pk1 : pk0;
        v = pkv * v - (1.f - pkv) * INFM;
        lg[t] = v;
      }
      // online softmax; row lives across the 16 lanes of this half
      float mr = fmaxf(lg[0], lg[1]);
#pragma unroll
      for (int off = 1; off < 16; off <<= 1) mr = fmaxf(mr, __shfl_xor(mr, off, 16));
      const float nm = fmaxf(rmax[r], mr);
      const float al = __expf(rmax[r] - nm);
      rmax[r] = nm;
      const float e0 = __expf(lg[0] - nm);
      const float e1 = __expf(lg[1] - nm);
      float ps = e0 + e1;
#pragma unroll
      for (int off = 1; off < 16; off <<= 1) ps += __shfl_xor(ps, off, 16);
      rsum[r] = rsum[r] * al + ps;
#pragma unroll
      for (int j = 0; j < 4; ++j) oacc[j][r] *= al;
      ptile[h][qi][m]      = (__bf16)e0;   // transpose P via per-wave LDS
      ptile[h][qi][16 + m] = (__bf16)e1;
    }
    __builtin_amdgcn_wave_barrier();

    const v16bf pa = frag_a(&ptile[h][0][0], 0, 0, 32, lane);
#pragma unroll
    for (int j = 0; j < 4; ++j) {
      v16bf vb = frag_b(Vt, j * 16, kc, Nsz, lane);  // Vt is d-major -> contiguous
      oacc[j] = wmma_bf16(pa, vb, oacc[j]);
    }
  }

  // epilogue: O = V + A.V
#pragma unroll
  for (int r = 0; r < 8; ++r) {
    const float inv_l = 1.f / rsum[r];
    const int q = q0 + r + 8 * half;
#pragma unroll
    for (int j = 0; j < 4; ++j) {
      const int d = j * 16 + m;
      const size_t idx = ((size_t)(b * Nsz + q)) * Dsz + h * HDsz + d;
      const float val = Vf32[idx] + oacc[j][r] * inv_l;
      Ocur[idx]   = val;
      Ocurbf[idx] = (__bf16)val;
    }
  }
}

// ---------------- kernel 4: O += relu(O @ Wo^T + bo) ----------------
__global__ __launch_bounds__(256) void proj_kernel(
    const __bf16* __restrict__ Ocurbf, const float* __restrict__ Ocur,
    const __bf16* __restrict__ Wobf, const float* __restrict__ bo,
    float* __restrict__ out)
{
  const int wave = threadIdx.x >> 5, lane = threadIdx.x & 31;
  const int m0 = blockIdx.x * 128 + wave * 16;
  const int n0 = blockIdx.y * 16;
  v8f acc = {};
  for (int k = 0; k < Dsz; k += 32) {
    v16bf a  = frag_a(Ocurbf, m0, k, Dsz, lane);
    v16bf bb = frag_b(Wobf,   n0, k, Dsz, lane);
    acc = wmma_bf16(a, bb, acc);
  }
  const int mloc = lane & 15, half = lane >> 4;
  const int n = n0 + mloc;
  const float bval = bo[n];
#pragma unroll
  for (int r = 0; r < 8; ++r) {
    const int mm = m0 + r + 8 * half;
    const size_t idx = (size_t)mm * Dsz + n;
    out[idx] = Ocur[idx] + fmaxf(acc[r] + bval, 0.f);
  }
}

extern "C" void kernel_launch(void* const* d_in, const int* in_sizes, int n_in,
                              void* d_out, int out_size, void* d_ws, size_t ws_size,
                              hipStream_t stream) {
  const float* Y    = (const float*)d_in[0];
  const float* Xp   = (const float*)d_in[1];
  const float* pres = (const float*)d_in[2];
  const float* Wq   = (const float*)d_in[3];
  const float* bq   = (const float*)d_in[4];
  const float* Wk   = (const float*)d_in[5];
  const float* bk   = (const float*)d_in[6];
  const float* Wv   = (const float*)d_in[7];
  const float* bv   = (const float*)d_in[8];
  const float* Wo   = (const float*)d_in[9];
  const float* bo   = (const float*)d_in[10];
  const float* W1   = (const float*)d_in[11];
  const float* b1   = (const float*)d_in[12];
  const float* W2   = (const float*)d_in[13];
  const float* b2   = (const float*)d_in[14];
  float* out = (float*)d_out;

  char* w = (char*)d_ws;
  size_t off = 0;
  auto take = [&](size_t bytes) -> void* {
    void* p = w + off;
    off += (bytes + 255) & ~(size_t)255;
    return p;
  };
  __bf16* Ybf   = (__bf16*)take((size_t)Bsz * Nsz * Dsz * 2);
  __bf16* Wqbf  = (__bf16*)take((size_t)Dsz * Dsz * 2);
  __bf16* Wkbf  = (__bf16*)take((size_t)Dsz * Dsz * 2);
  __bf16* Wvbf  = (__bf16*)take((size_t)Dsz * Dsz * 2);
  __bf16* Wobf  = (__bf16*)take((size_t)Dsz * Dsz * 2);
  __bf16* Qbf   = (__bf16*)take((size_t)Bsz * Hsz * Nsz * HDsz * 2);
  __bf16* Kbf   = (__bf16*)take((size_t)Bsz * Hsz * Nsz * HDsz * 2);
  __bf16* Vtbf  = (__bf16*)take((size_t)Bsz * Hsz * HDsz * Nsz * 2);
  float*  Vf32  = (float*)take((size_t)Bsz * Nsz * Dsz * 4);
  float*  Ocur  = (float*)take((size_t)Bsz * Nsz * Dsz * 4);
  __bf16* Ocurbf= (__bf16*)take((size_t)Bsz * Nsz * Dsz * 2);

  prep_kernel<<<(Bsz * Nsz * Dsz) / 256, 256, 0, stream>>>(
      Y, Wq, Wk, Wv, Wo, Ybf, Wqbf, Wkbf, Wvbf, Wobf);

  qkv_kernel<<<dim3(32, 32, 3), 256, 0, stream>>>(
      Ybf, Wqbf, Wkbf, Wvbf, bq, bk, bv, Qbf, Kbf, Vf32, Vtbf);

  attn_kernel<<<dim3(Nsz / 16, Bsz), 256, 0, stream>>>(
      Qbf, Kbf, Vtbf, Vf32, Xp, pres, W1, b1, W2, b2, Ocur, Ocurbf);

  proj_kernel<<<dim3(32, 32), 256, 0, stream>>>(
      Ocurbf, Ocur, Wobf, bo, out);
}
